// Attention_11407433138456
// MI455X (gfx1250) — matrix operand order repaired and checked by reading, not compiled, not run on hardware
//
#include <hip/hip_runtime.h>
#include <hip/hip_bf16.h>
#include <stdint.h>

// Attention fwd: B=4, Q=K=2048, D=1024, H=16, DH=64
#define BQ   4
#define QL   2048
#define KLEN 2048
#define DM   1024
#define HH   16
#define DHD  64
#define NOUT 1024   // H*DH

typedef __attribute__((ext_vector_type(16))) __bf16          v16bf;
typedef __attribute__((ext_vector_type(2)))  __bf16          v2bf;
typedef __attribute__((ext_vector_type(16))) unsigned short  v16u;
typedef __attribute__((ext_vector_type(8)))  unsigned short  v8u;
typedef __attribute__((ext_vector_type(8)))  unsigned int    v8i;
typedef __attribute__((ext_vector_type(8)))  float           v8f;

// bf16 WMMA fragment: 8 VGPRs, assembled via 16B halves or packed 32-bit words
union Frag {
    v16u v;
    v8u  h[2];
    v8i  w;
};

// native fptrunc -> v_cvt_pk_bf16_f32 on gfx12-family targets
__device__ __forceinline__ unsigned pkbf(float lo, float hi) {
    v2bf b;
    b[0] = (__bf16)lo;
    b[1] = (__bf16)hi;
    return __builtin_bit_cast(unsigned, b);
}
__device__ __forceinline__ unsigned short bf1(float f) {
    return __builtin_bit_cast(unsigned short, (__bf16)f);
}

__device__ __forceinline__ v8f vzero8() {
    v8f z;
#pragma unroll
    for (int i = 0; i < 8; ++i) z[i] = 0.0f;
    return z;
}

// D = A(16x32 bf16) * B(32x16 bf16) + C(16x16 f32), wave32 WMMA
__device__ __forceinline__ v8f wmma_bf16(v16u a, v16u b, v8f c) {
    return __builtin_amdgcn_wmma_f32_16x16x32_bf16(
        false, __builtin_bit_cast(v16bf, a),
        false, __builtin_bit_cast(v16bf, b),
        (short)0, c, false, false);
}

// ---------------------------------------------------------------------------
// Kernel 1: Q/K/V projection. grid=(64,16,3), block=256 (8 waves).
// Per k-step the block cooperatively stages the shared 32(d)x64(n) W tile
// into LDS transposed as [n][d] bf16; B fragments are two ds_load_b128 each.
// q,k stored [B,H,Q,DH]; v stored transposed [B,H,DH,K] for the P*V B-layout.
// ---------------------------------------------------------------------------
__global__ void __launch_bounds__(256) qkv_proj_kernel(
    const float* __restrict__ x,
    const float* __restrict__ Wq, const float* __restrict__ Wk,
    const float* __restrict__ Wv,
    unsigned short* __restrict__ q_ws, unsigned short* __restrict__ k_ws,
    unsigned short* __restrict__ v_ws)
{
    __shared__ unsigned short w_t[64 * 32];   // [n_local][d_local] bf16, 4 KB

    const int tid  = threadIdx.x;
    const int lane = tid & 31;
    const int wid  = tid >> 5;
    const int half = lane >> 4;
    const int l15  = lane & 15;
    const int z    = blockIdx.z;
    const float* W = (z == 0) ? Wq : (z == 1) ? Wk : Wv;
    unsigned short* outp = (z == 0) ? q_ws : (z == 1) ? k_ws : v_ws;

    const int m0 = (blockIdx.x * 8 + wid) * 16;
    const int n0 = blockIdx.y * 64;

    // staging assignment: thread -> (d_local, 8-wide n chunk)
    const int sd = tid >> 3;          // 0..31
    const int sn = (tid & 7) * 8;     // 0,8,..,56

    v8f acc[4];
#pragma unroll
    for (int t = 0; t < 4; ++t) acc[t] = vzero8();

    const float* xr = x + (size_t)(m0 + l15) * DM;

    for (int kk = 0; kk < DM; kk += 32) {
        // ---- stage W tile (coalesced rows, store transposed bf16)
        {
            const float* wr = W + (size_t)(kk + sd) * NOUT + n0 + sn;
            float4 w0 = *(const float4*)(wr);
            float4 w1 = *(const float4*)(wr + 4);
            unsigned short* dst = w_t + sd;           // column d_local
            dst[(sn + 0) * 32] = bf1(w0.x);
            dst[(sn + 1) * 32] = bf1(w0.y);
            dst[(sn + 2) * 32] = bf1(w0.z);
            dst[(sn + 3) * 32] = bf1(w0.w);
            dst[(sn + 4) * 32] = bf1(w1.x);
            dst[(sn + 5) * 32] = bf1(w1.y);
            dst[(sn + 6) * 32] = bf1(w1.z);
            dst[(sn + 7) * 32] = bf1(w1.w);
        }
        __syncthreads();

        if (kk + 32 < DM) __builtin_prefetch(xr + kk + 32, 0, 3);

        // A fragment (16x32): lane=row; elems 0..7 -> d=kk+half*8+j,
        // elems 8..15 -> d=kk+16+half*8+j
        const int da = kk + half * 8;
        float4 a0 = *(const float4*)(xr + da);
        float4 a1 = *(const float4*)(xr + da + 4);
        float4 a2 = *(const float4*)(xr + da + 16);
        float4 a3 = *(const float4*)(xr + da + 20);
        Frag av;
        av.w[0] = pkbf(a0.x, a0.y); av.w[1] = pkbf(a0.z, a0.w);
        av.w[2] = pkbf(a1.x, a1.y); av.w[3] = pkbf(a1.z, a1.w);
        av.w[4] = pkbf(a2.x, a2.y); av.w[5] = pkbf(a2.z, a2.w);
        av.w[6] = pkbf(a3.x, a3.y); av.w[7] = pkbf(a3.z, a3.w);

        // B fragments (32x16): preload all 4, then 4 back-to-back WMMAs
        Frag bv[4];
#pragma unroll
        for (int t = 0; t < 4; ++t) {
            const unsigned short* wp = w_t + (t * 16 + l15) * 32 + half * 8;
            bv[t].h[0] = *(const v8u*)(wp);
            bv[t].h[1] = *(const v8u*)(wp + 16);
        }
#pragma unroll
        for (int t = 0; t < 4; ++t)
            acc[t] = wmma_bf16(av.v, bv[t].v, acc[t]);
        __syncthreads();
    }

    // C layout: lane l holds rows r + 8*(l>>4), col = l&15 (per 16-wide tile)
#pragma unroll
    for (int t = 0; t < 4; ++t) {
#pragma unroll
        for (int r = 0; r < 8; ++r) {
            const int m  = m0 + r + 8 * half;
            const int n  = n0 + t * 16 + l15;
            const int b  = m >> 11, qi = m & 2047;
            const int h  = n >> 6,  dh = n & 63;
            size_t addr;
            if (z == 2)
                addr = (((size_t)b * HH + h) * DHD + dh) * (size_t)KLEN + qi;
            else
                addr = (((size_t)b * HH + h) * (size_t)QL + qi) * DHD + dh;
            outp[addr] = bf1(acc[t][r]);
        }
    }
}

// ---------------------------------------------------------------------------
// Kernel 2: flash attention. grid=(32, B*H), block=128 (4 waves).
// Double-buffered K/V tiles staged with CDNA5 async global->LDS copies
// (ASYNCcnt): tile k+1 DMA overlaps WMMA compute on tile k.
// ---------------------------------------------------------------------------
__global__ void __launch_bounds__(128) attn_kernel(
    const unsigned short* __restrict__ q_ws,
    const unsigned short* __restrict__ k_ws,
    const unsigned short* __restrict__ v_ws,
    const unsigned char*  __restrict__ mask,   // JAX bool -> 1 byte/elem
    unsigned short* __restrict__ a_ws)         // [B, Q, H*DH] bf16
{
    __shared__ unsigned short k_tile[2][32 * 64];   // [k][d]  2x 4 KB
    __shared__ unsigned short v_tile[2][64 * 32];   // [d][k]  2x 4 KB
    __shared__ unsigned short lds_p[4][16 * 32];    // per-wave P staging 4 KB

    const int tid  = threadIdx.x;
    const int lane = tid & 31;
    const int wid  = tid >> 5;
    const int half = lane >> 4;
    const int l15  = lane & 15;
    const int bh   = blockIdx.y;
    const int b    = bh >> 4, h = bh & 15;
    const int q0   = blockIdx.x * 64 + wid * 16;

    // async-stage one 32-k tile (K: contiguous 4 KB; V: 64 rows x 64 B)
    auto stage = [&](int k0, int buf) {
        const unsigned kb = (unsigned)(uintptr_t)(void*)&k_tile[buf][0];
        const unsigned vb = (unsigned)(uintptr_t)(void*)&v_tile[buf][0];
        const unsigned long long gk =
            (unsigned long long)(uintptr_t)(k_ws + ((size_t)bh * KLEN + k0) * DHD)
            + (unsigned)tid * 16u;
        const unsigned lk = kb + (unsigned)tid * 16u;
        asm volatile("global_load_async_to_lds_b128 %0, %1, off"
                     :: "v"(lk), "v"(gk) : "memory");
        asm volatile("global_load_async_to_lds_b128 %0, %1, off"
                     :: "v"(lk + 2048u), "v"(gk + 2048ull) : "memory");
#pragma unroll
        for (int rep = 0; rep < 2; ++rep) {
            const int u = tid + rep * 128;           // 16B unit id (0..255)
            const int d = u >> 2, part = u & 3;
            const unsigned long long gv =
                (unsigned long long)(uintptr_t)(v_ws + ((size_t)bh * DHD + d) * KLEN + k0)
                + (unsigned)part * 16u;
            const unsigned lv = vb + (unsigned)u * 16u;
            asm volatile("global_load_async_to_lds_b128 %0, %1, off"
                         :: "v"(lv), "v"(gv) : "memory");
        }
    };

    // Q fragments: 16 rows x DH=64 -> two 16x32 A fragments
    Frag qa[2];
    {
        const unsigned short* qr =
            q_ws + ((size_t)bh * QL + (q0 + l15)) * DHD + half * 8;
#pragma unroll
        for (int f = 0; f < 2; ++f) {
            qa[f].h[0] = *(const v8u*)(qr + f * 32);
            qa[f].h[1] = *(const v8u*)(qr + f * 32 + 16);
        }
    }

    float mrow[8], lrow[8];
#pragma unroll
    for (int r = 0; r < 8; ++r) { mrow[r] = -3.0e38f; lrow[r] = 0.0f; }
    v8f o[4];
#pragma unroll
    for (int t = 0; t < 4; ++t) o[t] = vzero8();

    const float scale = 0.125f;            // DH^-0.5
    unsigned short* pl = lds_p[wid];

    // prologue: stage tile 0
    stage(0, 0);
    asm volatile("s_wait_asynccnt 0x0" ::: "memory");
    __syncthreads();

    int buf = 0;
    for (int k0 = 0; k0 < KLEN; k0 += 32) {
        // kick off DMA for the next tile into the other buffer
        if (k0 + 32 < KLEN) stage(k0 + 32, buf ^ 1);

        const unsigned short* kt = k_tile[buf];
        const unsigned short* vt = v_tile[buf];

        // ---- S = Q K^T : two 16x16 tiles, each 2 WMMAs over d (from LDS)
        Frag kbf[2][2];
#pragma unroll
        for (int g = 0; g < 2; ++g) {
            const unsigned short* kp = kt + (g * 16 + l15) * 64 + half * 8;
            kbf[g][0].h[0] = *(const v8u*)(kp);
            kbf[g][0].h[1] = *(const v8u*)(kp + 16);
            kbf[g][1].h[0] = *(const v8u*)(kp + 32);
            kbf[g][1].h[1] = *(const v8u*)(kp + 48);
        }
        v8f s[2];
#pragma unroll
        for (int g = 0; g < 2; ++g) {
            s[g] = wmma_bf16(qa[0].v, kbf[g][0].v, vzero8());
            s[g] = wmma_bf16(qa[1].v, kbf[g][1].v, s[g]);
        }

        // ---- mask + online softmax in C-layout (lane owns 8 rows x 2 cols)
        float p0[8], p1[8];
        const int qmBase = q0 + 8 * half;
#pragma unroll
        for (int r = 0; r < 8; ++r) {
            const unsigned char* mp =
                mask + ((size_t)b * QL + (qmBase + r)) * KLEN + k0 + l15;
            const bool ok0 = mp[0]  != 0;
            const bool ok1 = mp[16] != 0;
            float s0 = ok0 ? s[0][r] * scale : -3.0e38f;
            float s1 = ok1 ? s[1][r] * scale : -3.0e38f;
            float v = fmaxf(s0, s1);
#pragma unroll
            for (int i = 1; i < 16; i <<= 1) v = fmaxf(v, __shfl_xor(v, i, 32));
            const float newm  = fmaxf(mrow[r], v);
            const float alpha = __expf(mrow[r] - newm);
            const float e0 = ok0 ? __expf(s0 - newm) : 0.0f;
            const float e1 = ok1 ? __expf(s1 - newm) : 0.0f;
            float rs = e0 + e1;
#pragma unroll
            for (int i = 1; i < 16; i <<= 1) rs += __shfl_xor(rs, i, 32);
            lrow[r] = lrow[r] * alpha + rs;
            mrow[r] = newm;
            p0[r] = e0; p1[r] = e1;
#pragma unroll
            for (int t = 0; t < 4; ++t) o[t][r] *= alpha;
        }

        // ---- C-layout -> A-layout via per-wave LDS bounce
#pragma unroll
        for (int r = 0; r < 8; ++r) {
            const int mr = r + 8 * half;
            pl[mr * 32 + l15]      = bf1(p0[r]);
            pl[mr * 32 + 16 + l15] = bf1(p1[r]);
        }
        asm volatile("s_wait_dscnt 0x0" ::: "memory");
        Frag pa;
        {
            const unsigned short* pr = pl + l15 * 32 + half * 8;
            pa.h[0] = *(const v8u*)(pr);
            pa.h[1] = *(const v8u*)(pr + 16);
        }

        // ---- O += P(16x32) * V(32x64); V tile is [d][k] in LDS
        Frag vb[4];
#pragma unroll
        for (int t = 0; t < 4; ++t) {
            const unsigned short* vp = vt + (t * 16 + l15) * 32 + half * 8;
            vb[t].h[0] = *(const v8u*)(vp);
            vb[t].h[1] = *(const v8u*)(vp + 16);
        }
#pragma unroll
        for (int t = 0; t < 4; ++t)
            o[t] = wmma_bf16(pa.v, vb[t].v, o[t]);

        // next-tile DMA must be complete and everyone done with this buffer
        asm volatile("s_wait_asynccnt 0x0" ::: "memory");
        __syncthreads();
        buf ^= 1;
    }

    // ---- normalize + store bf16 [B, Q, H*DH]
#pragma unroll
    for (int r = 0; r < 8; ++r) {
        const float inv = 1.0f / fmaxf(lrow[r], 1e-9f);
        const int qm = q0 + r + 8 * half;
        unsigned short* op =
            a_ws + ((size_t)(b * QL + qm)) * NOUT + h * DHD + l15;
#pragma unroll
        for (int t = 0; t < 4; ++t)
            op[t * 16] = bf1(o[t][r] * inv);
    }
}

// ---------------------------------------------------------------------------
// Kernel 3: out = attn @ Wo + bo. grid=(64,16), block=256 (8 waves).
// Same LDS-staged transposed Wo tile as kernel 1; A is already bf16.
// ---------------------------------------------------------------------------
__global__ void __launch_bounds__(256) out_proj_kernel(
    const unsigned short* __restrict__ a_ws,
    const float* __restrict__ Wo,
    const float* __restrict__ bo,
    float* __restrict__ out)
{
    __shared__ unsigned short w_t[64 * 32];   // [n_local][d_local] bf16, 4 KB

    const int tid  = threadIdx.x;
    const int lane = tid & 31;
    const int wid  = tid >> 5;
    const int half = lane >> 4;
    const int l15  = lane & 15;
    const int m0 = (blockIdx.x * 8 + wid) * 16;
    const int n0 = blockIdx.y * 64;

    const int sd = tid >> 3;
    const int sn = (tid & 7) * 8;

    v8f acc[4];
#pragma unroll
    for (int t = 0; t < 4; ++t) acc[t] = vzero8();

    const unsigned short* ar = a_ws + (size_t)(m0 + l15) * DM + half * 8;

    for (int kk = 0; kk < DM; kk += 32) {
        // ---- stage Wo tile (coalesced rows, store transposed bf16)
        {
            const float* wr = Wo + (size_t)(kk + sd) * DM + n0 + sn;
            float4 w0 = *(const float4*)(wr);
            float4 w1 = *(const float4*)(wr + 4);
            unsigned short* dst = w_t + sd;
            dst[(sn + 0) * 32] = bf1(w0.x);
            dst[(sn + 1) * 32] = bf1(w0.y);
            dst[(sn + 2) * 32] = bf1(w0.z);
            dst[(sn + 3) * 32] = bf1(w0.w);
            dst[(sn + 4) * 32] = bf1(w1.x);
            dst[(sn + 5) * 32] = bf1(w1.y);
            dst[(sn + 6) * 32] = bf1(w1.z);
            dst[(sn + 7) * 32] = bf1(w1.w);
        }
        __syncthreads();

        if (kk + 32 < DM) __builtin_prefetch(ar + kk + 32, 0, 3);

        Frag av;
        av.h[0] = *(const v8u*)(ar + kk);
        av.h[1] = *(const v8u*)(ar + kk + 16);

        Frag bv[4];
#pragma unroll
        for (int t = 0; t < 4; ++t) {
            const unsigned short* wp = w_t + (t * 16 + l15) * 32 + half * 8;
            bv[t].h[0] = *(const v8u*)(wp);
            bv[t].h[1] = *(const v8u*)(wp + 16);
        }
#pragma unroll
        for (int t = 0; t < 4; ++t)
            acc[t] = wmma_bf16(av.v, bv[t].v, acc[t]);
        __syncthreads();
    }

#pragma unroll
    for (int t = 0; t < 4; ++t) {
#pragma unroll
        for (int r = 0; r < 8; ++r) {
            const int m = m0 + r + 8 * half;
            const int n = n0 + t * 16 + l15;
            out[(size_t)m * DM + n] = acc[t][r] + bo[n];
        }
    }
}

// ---------------------------------------------------------------------------
extern "C" void kernel_launch(void* const* d_in, const int* in_sizes, int n_in,
                              void* d_out, int out_size, void* d_ws, size_t ws_size,
                              hipStream_t stream)
{
    const float*         x    = (const float*)d_in[0];
    const unsigned char* mask = (const unsigned char*)d_in[1];  // JAX bool
    const float*         Wq   = (const float*)d_in[2];
    const float*         Wk   = (const float*)d_in[3];
    const float*         Wv   = (const float*)d_in[4];
    const float*         Wo   = (const float*)d_in[5];
    const float*         bo   = (const float*)d_in[6];
    float*               out  = (float*)d_out;

    // workspace: q,k,v,attn each B*H*Q*DH bf16 = 16 MB -> 64 MB total
    const size_t elems = (size_t)BQ * HH * QL * DHD;   // 8388608
    unsigned short* q_ws = (unsigned short*)d_ws;
    unsigned short* k_ws = q_ws + elems;
    unsigned short* v_ws = k_ws + elems;
    unsigned short* a_ws = v_ws + elems;

    dim3 blk1(256), grd1(64, 16, 3);
    qkv_proj_kernel<<<grd1, blk1, 0, stream>>>(x, Wq, Wk, Wv, q_ws, k_ws, v_ws);

    dim3 blk2(128), grd2(32, BQ * HH);
    attn_kernel<<<grd2, blk2, 0, stream>>>(q_ws, k_ws, v_ws, mask, a_ws);

    dim3 blk3(256), grd3(64, 16);
    out_proj_kernel<<<grd3, blk3, 0, stream>>>(a_ws, Wo, bo, out);
}
